// GMMConv_26216480375296
// MI455X (gfx1250) — matrix operand order repaired and checked by reading, not compiled, not run on hardware
//
#include <hip/hip_runtime.h>
#include <hip/hip_bf16.h>

// GMMConv for MI455X (gfx1250), wave32 + WMMA.
//
// Inputs (setup_inputs order):
//  0 feat      [N, 64]  f32
//  1 pseudo    [E, 3]   f32
//  2 src       [E]      i32
//  3 dst       [E]      i32
//  4 W_fc      [256,64] f32   (K*out rows, in_feats cols)
//  5 mu        [4, 3]   f32
//  6 inv_sigma [4, 3]   f32
//  7 bias      [64]     f32
// Output: rst [N, 64] f32
// Scratch: h = feat @ W_fc.T  -> [N, 256] f32 in d_ws (102.4 MB, L2-resident).

typedef __attribute__((ext_vector_type(16))) _Float16 v16h;
typedef __attribute__((ext_vector_type(8)))  float    v8f;

#define IN_FEATS_C  64
#define OUT_FEATS_C 64
#define NKERN_C     4
#define KOUT_C      (NKERN_C * OUT_FEATS_C)   // 256

// ---------------------------------------------------------------------------
// Kernel 1: h = feat @ W_fc^T via v_wmma_f32_16x16x32_f16.
// One wave computes one 16(node) x 16(out-col) tile. blockDim = 256 (8 waves),
// grid = (ceil(N/16), 2): blockIdx.y*8 + waveId selects one of 16 column tiles.
// ---------------------------------------------------------------------------
__global__ __launch_bounds__(256)
void gmm_project_wmma(const float* __restrict__ feat,
                      const float* __restrict__ Wfc,
                      float* __restrict__ h,
                      int n_nodes)
{
    const unsigned lane = threadIdx.x & 31u;
    const unsigned wid  = threadIdx.x >> 5;          // 0..7
    const unsigned hi   = lane >> 4;                 // lane[4]: K-half select
    const unsigned m    = lane & 15u;                // row (A) / col (B/C)

    const int row0 = blockIdx.x * 16;                // node tile base
    const int col0 = (blockIdx.y * 8 + wid) * 16;    // output-column tile base

    if (row0 >= n_nodes) return;                     // uniform per wave

    // Clamp A-row for ragged final tile (stores are guarded below).
    const int arowIdx = (row0 + (int)m < n_nodes) ? (row0 + (int)m) : row0;
    const float* arow = feat + (size_t)arowIdx * IN_FEATS_C;
    const float* brow = Wfc  + (size_t)(col0 + (int)m) * IN_FEATS_C;

    // --- A operand: 16x32 f16 layout ---------------------------------------
    // v16h elems [0..7]  -> VGPR0..3 : K = kk + hi*8  + {0..7}
    // v16h elems [8..15] -> VGPR4..7 : K = kk + 16 + hi*8 + {0..7}
    v16h a0, a1;
    #pragma unroll
    for (int i = 0; i < 8; ++i) {
        a0[i]     = (_Float16)arow[ 0 + hi * 8 + i];
        a0[8 + i] = (_Float16)arow[16 + hi * 8 + i];
        a1[i]     = (_Float16)arow[32 + hi * 8 + i];
        a1[8 + i] = (_Float16)arow[48 + hi * 8 + i];
    }

    // --- B operand: 32x16 f16 layout ---------------------------------------
    // elems [0..7]: K = kk + hi*8 + {0..7}; elems [8..15]: K = kk + 16 + hi*8 + {0..7} (N=lane%16).
    // B[k][n] = W_fc[col0+n][k] -> contiguous along k in W_fc rows.
    v16h b0, b1;
    #pragma unroll
    for (int i = 0; i < 8; ++i) {
        b0[i]     = (_Float16)brow[ 0 + hi * 8 + i];
        b0[8 + i] = (_Float16)brow[16 + hi * 8 + i];
        b1[i]     = (_Float16)brow[32 + hi * 8 + i];
        b1[8 + i] = (_Float16)brow[48 + hi * 8 + i];
    }

    // --- D = A*B + C, f32 accumulate, K = 64 via two x32 WMMAs -------------
    v8f c = {};
    c = __builtin_amdgcn_wmma_f32_16x16x32_f16(false, a0, false, b0,
                                               (short)0, c, false, false);
    c = __builtin_amdgcn_wmma_f32_16x16x32_f16(false, a1, false, b1,
                                               (short)0, c, false, false);

    // --- C/D store: VGPR v -> row (v + hi*8), col = m ----------------------
    float* crow = h + (size_t)row0 * KOUT_C + col0 + m;
    if (row0 + 16 <= n_nodes) {
        // Uniform fast path: full tile, no per-lane guards / EXEC churn.
        #pragma unroll
        for (int v = 0; v < 8; ++v)
            crow[(size_t)(v + hi * 8) * KOUT_C] = c[v];
    } else {
        // Ragged final tile (not hit for N % 16 == 0).
        #pragma unroll
        for (int v = 0; v < 8; ++v) {
            const int r = v + (int)hi * 8;
            if (row0 + r < n_nodes)
                crow[(size_t)r * KOUT_C] = c[v];
        }
    }
}

// ---------------------------------------------------------------------------
// Kernel 2: out = feat + bias  (residual + bias; also initializes d_out)
// float4-vectorized: gid indexes groups of 4 feats.
// ---------------------------------------------------------------------------
__global__ __launch_bounds__(256)
void gmm_init(const float* __restrict__ feat,
              const float* __restrict__ bias,
              float* __restrict__ out,
              int total4)
{
    int gid = blockIdx.x * blockDim.x + threadIdx.x;
    if (gid >= total4) return;
    const int o4 = (gid & 15) << 2;   // feature offset within node (0,4,...,60)
    float4 f = *(const float4*)(feat + (size_t)gid * 4);
    float4 b = *(const float4*)(bias + o4);
    float4 r = make_float4(f.x + b.x, f.y + b.y, f.z + b.z, f.w + b.w);
    *(float4*)(out + (size_t)gid * 4) = r;
}

// ---------------------------------------------------------------------------
// Kernel 3: edge scatter. 16 threads per edge, 4 output feats per thread.
// g[e,k] recomputed per thread (cheap; mu/inv_sigma reads are wave-uniform
// and scalarize). h[src] float4 gathers hit the 192MB L2 (h is 102.4MB);
// non-returning global_atomic_add_f32 into out[dst].
// ---------------------------------------------------------------------------
__global__ __launch_bounds__(256)
void gmm_scatter(const float* __restrict__ h,
                 const float* __restrict__ pseudo,
                 const int* __restrict__ src,
                 const int* __restrict__ dst,
                 const float* __restrict__ mu,
                 const float* __restrict__ inv_sigma,
                 float* __restrict__ out,
                 int n_edges)
{
    long gid = (long)blockIdx.x * blockDim.x + threadIdx.x;
    int e  = (int)(gid >> 4);
    int o4 = ((int)gid & 15) << 2;
    if (e >= n_edges) return;

    const float p0 = pseudo[(size_t)e * 3 + 0];
    const float p1 = pseudo[(size_t)e * 3 + 1];
    const float p2 = pseudo[(size_t)e * 3 + 2];

    const float* hs = h + (size_t)src[e] * KOUT_C + o4;

    float4 acc = make_float4(0.f, 0.f, 0.f, 0.f);
    #pragma unroll
    for (int k = 0; k < NKERN_C; ++k) {
        const float d0 = p0 - mu[k * 3 + 0];
        const float d1 = p1 - mu[k * 3 + 1];
        const float d2 = p2 - mu[k * 3 + 2];
        const float s0 = inv_sigma[k * 3 + 0];
        const float s1 = inv_sigma[k * 3 + 1];
        const float s2 = inv_sigma[k * 3 + 2];
        const float g = __expf(-0.5f * (d0 * d0 * s0 * s0 +
                                        d1 * d1 * s1 * s1 +
                                        d2 * d2 * s2 * s2));
        float4 hv = *(const float4*)(hs + k * OUT_FEATS_C);
        acc.x += hv.x * g;
        acc.y += hv.y * g;
        acc.z += hv.z * g;
        acc.w += hv.w * g;
    }

    float* op = out + (size_t)dst[e] * OUT_FEATS_C + o4;
    atomicAdd(op + 0, acc.x);
    atomicAdd(op + 1, acc.y);
    atomicAdd(op + 2, acc.z);
    atomicAdd(op + 3, acc.w);
}

// ---------------------------------------------------------------------------
extern "C" void kernel_launch(void* const* d_in, const int* in_sizes, int n_in,
                              void* d_out, int out_size, void* d_ws, size_t ws_size,
                              hipStream_t stream)
{
    const float* feat      = (const float*)d_in[0];
    const float* pseudo    = (const float*)d_in[1];
    const int*   src       = (const int*)d_in[2];
    const int*   dst       = (const int*)d_in[3];
    const float* Wfc       = (const float*)d_in[4];
    const float* mu        = (const float*)d_in[5];
    const float* inv_sigma = (const float*)d_in[6];
    const float* bias      = (const float*)d_in[7];
    float*       out       = (float*)d_out;
    float*       h         = (float*)d_ws;   // [N, 256] f32

    const int n_nodes = in_sizes[0] / IN_FEATS_C;
    const int n_edges = in_sizes[2];

    // 1) h = feat @ W_fc^T  (WMMA)
    {
        dim3 grid((n_nodes + 15) / 16, 2);
        gmm_project_wmma<<<grid, 256, 0, stream>>>(feat, Wfc, h, n_nodes);
    }

    // 2) out = feat + bias
    {
        const int total4 = n_nodes * (OUT_FEATS_C / 4);
        gmm_init<<<(total4 + 255) / 256, 256, 0, stream>>>(feat, bias, out, total4);
    }

    // 3) scatter-add messages
    {
        const long threads = (long)n_edges * 16;
        const int  blocks  = (int)((threads + 255) / 256);
        gmm_scatter<<<blocks, 256, 0, stream>>>(h, pseudo, src, dst,
                                                mu, inv_sigma, out, n_edges);
    }
}
